// DecoderBlock_50714973831866
// MI455X (gfx1250) — matrix operand order repaired and checked by reading, not compiled, not run on hardware
//
#include <hip/hip_runtime.h>
#include <hip/hip_bf16.h>

typedef __bf16 bf16_t;
typedef __attribute__((ext_vector_type(16))) __bf16 v16bf;
typedef __attribute__((ext_vector_type(8)))  float  v8f;

#define NEGBIG (-1.0e30f)

// ---- gfx1250 async global->LDS copy (ASYNCcnt-tracked, no VGPR round trip) ----
__device__ __forceinline__ void async_ld_b128(const void* gptr, void* lptr) {
    unsigned l = (unsigned)(uintptr_t)lptr;   // low 32 bits of generic ptr = LDS offset
    asm volatile("global_load_async_to_lds_b128 %0, %1, off"
                 :: "v"(l), "v"(gptr) : "memory");
}
__device__ __forceinline__ void wait_async0() {
    asm volatile("s_wait_asynccnt 0x0" ::: "memory");
}

// ---------------------------------------------------------------- cast f32->bf16
__global__ __launch_bounds__(256) void cast_k(const float* __restrict__ in,
                                              bf16_t* __restrict__ out, long n) {
    long i = (long)blockIdx.x * 256 + threadIdx.x;
    long stride = (long)gridDim.x * 256;
    for (; i < n; i += stride) out[i] = (bf16_t)in[i];
}

// ---------------------------------------------------------------- LayerNorm -> bf16
__global__ __launch_bounds__(256) void ln_k(const float* __restrict__ x,
                                            const float* __restrict__ g,
                                            const float* __restrict__ b,
                                            bf16_t* __restrict__ out, int C) {
    int row = blockIdx.x, tid = threadIdx.x;
    const float* xr = x + (long)row * C;
    float4 v = ((const float4*)xr)[tid];       // C=1024, 4 per thread
    float s = v.x + v.y + v.z + v.w;
    __shared__ float red[8];
    for (int m2 = 1; m2 < 32; m2 <<= 1) s += __shfl_xor(s, m2, 32);
    if ((tid & 31) == 0) red[tid >> 5] = s;
    __syncthreads();
    float mu = 0.f;
    for (int i = 0; i < 8; i++) mu += red[i];
    mu *= (1.0f / (float)C);
    float dx = v.x - mu, dy = v.y - mu, dz = v.z - mu, dw = v.w - mu;
    float s2 = dx * dx + dy * dy + dz * dz + dw * dw;
    for (int m2 = 1; m2 < 32; m2 <<= 1) s2 += __shfl_xor(s2, m2, 32);
    __syncthreads();
    if ((tid & 31) == 0) red[tid >> 5] = s2;
    __syncthreads();
    float var = 0.f;
    for (int i = 0; i < 8; i++) var += red[i];
    var *= (1.0f / (float)C);
    float inv = rsqrtf(var + 1e-5f);
    float4 gv = ((const float4*)g)[tid];
    float4 bv = ((const float4*)b)[tid];
    long o = (long)row * C + (long)tid * 4;
    out[o + 0] = (bf16_t)(dx * inv * gv.x + bv.x);
    out[o + 1] = (bf16_t)(dy * inv * gv.y + bv.y);
    out[o + 2] = (bf16_t)(dz * inv * gv.z + bv.z);
    out[o + 3] = (bf16_t)(dw * inv * gv.w + bv.w);
}

// ---------------------------------------------------------------- tiled WMMA GEMM
// BM=128, BK=32, BNT in {64,128}.  8 waves (4 x 2), each wave computes
// 32 x (BNT/2) output: 2 x FN frags, FN = BNT/32.
// A tile: async global->LDS, double buffered.  B tile: register-pipelined and
// stored transposed (Bt[n][k]) so B-frags are two contiguous ds_load_b128.
// EPI 0: outB = bf16(acc)              (QKV; per-z B/C strides)
// EPI 1: outB = bf16(relu(acc + bias)) (FF1)
// EPI 2: outF = acc + bias + res       (Wo + residual, FF2 + residual)
#define BM 128
#define BK 32

template <int EPI, int BNT>
__global__ __launch_bounds__(256) void gemm_bf16_k(
    const bf16_t* __restrict__ A, const bf16_t* __restrict__ Bm,
    int M, int N, int K,
    const float* __restrict__ bias, const float* __restrict__ res,
    float* __restrict__ outF, bf16_t* __restrict__ outB,
    long bStrideZ, long cStrideZ) {
    constexpr int FN  = BNT / 32;                 // B/N frags per wave
    constexpr int NB4 = BNT / 64;                 // B-stage uint4 per thread
    constexpr int BC  = BNT / 8;                  // uint4 per B row
    __shared__ __align__(16) bf16_t As[2][BM * BK];
    __shared__ __align__(16) bf16_t Bt[2][BNT * BK];   // transposed: [n][k]

    const int tid = threadIdx.x;
    const int wave = tid >> 5, lane = tid & 31;
    const int wm = wave >> 1, wn = wave & 1;      // waves 4x2 over (M,N)
    const int l16 = lane & 15, lhalf = lane >> 4;
    const int m0 = blockIdx.y * BM;
    const int n0 = blockIdx.x * BNT;
    Bm += (long)blockIdx.z * bStrideZ;

    const int ar = tid >> 2, ac = tid & 3;        // A stage: 2 uint4 / thread

    v8f acc[2][FN];
#pragma unroll
    for (int i = 0; i < 2; i++)
#pragma unroll
        for (int j = 0; j < FN; j++)
#pragma unroll
            for (int e = 0; e < 8; e++) acc[i][j][e] = 0.f;

    // ---- prologue: tile 0 in flight
#pragma unroll
    for (int i = 0; i < 2; i++) {
        int idx = tid + i * 256;
        async_ld_b128(A + (long)(m0 + ar + i * 64) * K + ac * 8, &As[0][idx * 8]);
    }
    uint4 gb[NB4];
#pragma unroll
    for (int i = 0; i < NB4; i++) {
        int idx = tid + i * 256;
        gb[i] = ((const uint4*)(Bm + (long)(idx / BC) * N + n0))[idx % BC];
    }

    const int ntiles = K / BK;
    for (int it = 0; it < ntiles; ++it) {
        const int cur = it & 1;
        {   // transposed B store: B[r][c*8+e] -> Bt[c*8+e][r]
#pragma unroll
            for (int i = 0; i < NB4; i++) {
                int idx = tid + i * 256;
                int r = idx / BC, c = idx % BC;
                const bf16_t* hb = (const bf16_t*)&gb[i];
#pragma unroll
                for (int e = 0; e < 8; e++) Bt[cur][(c * 8 + e) * BK + r] = hb[e];
            }
        }
        wait_async0();
        __syncthreads();

        if (it + 1 < ntiles) {                    // launch tile it+1
            const int k0n = (it + 1) * BK;
            const int nxt = (it + 1) & 1;
#pragma unroll
            for (int i = 0; i < 2; i++) {
                int idx = tid + i * 256;
                async_ld_b128(A + (long)(m0 + ar + i * 64) * K + k0n + ac * 8,
                              &As[nxt][idx * 8]);
            }
#pragma unroll
            for (int i = 0; i < NB4; i++) {
                int idx = tid + i * 256;
                gb[i] = ((const uint4*)(Bm + (long)(k0n + idx / BC) * N + n0))[idx % BC];
            }
        }

        // load all frags first (distinct regs -> batched ds, partial dscnt waits)
        v16bf af[2], bfr[FN];
#pragma unroll
        for (int i = 0; i < 2; i++) {             // A frag: 16x32 (ISA layout)
            int m = wm * 32 + i * 16 + l16;
            const bf16_t* ap = &As[cur][m * BK + lhalf * 8];
            ((uint4*)&af[i])[0] = *(const uint4*)ap;
            ((uint4*)&af[i])[1] = *(const uint4*)(ap + 16);
        }
#pragma unroll
        for (int j = 0; j < FN; j++) {            // B frag: 16x32 (ISA layout)
            int n = wn * (BNT / 2) + j * 16 + l16;
            const bf16_t* bp = &Bt[cur][n * BK + lhalf * 8];
            ((uint4*)&bfr[j])[0] = *(const uint4*)bp;
            ((uint4*)&bfr[j])[1] = *(const uint4*)(bp + 16);
        }
#pragma unroll
        for (int i = 0; i < 2; i++)
#pragma unroll
            for (int j = 0; j < FN; j++)
                acc[i][j] = __builtin_amdgcn_wmma_f32_16x16x32_bf16(
                    false, af[i], false, bfr[j], (short)0, acc[i][j], false, false);
    }

#pragma unroll
    for (int i = 0; i < 2; i++)
#pragma unroll
        for (int j = 0; j < FN; j++) {
            int col = n0 + wn * (BNT / 2) + j * 16 + l16;
#pragma unroll
            for (int v = 0; v < 8; v++) {
                int row = m0 + wm * 32 + i * 16 + v + lhalf * 8;
                float val = acc[i][j][v];
                if (EPI == 0) {
                    outB[(long)blockIdx.z * cStrideZ + (long)row * N + col] = (bf16_t)val;
                } else if (EPI == 1) {
                    val += bias[col];
                    val = fmaxf(val, 0.f);
                    outB[(long)row * N + col] = (bf16_t)val;
                } else {
                    val += bias[col] + res[(long)row * N + col];
                    outF[(long)row * N + col] = val;
                }
            }
        }
}

// ---------------------------------------------------------------- flash attention
// Block: 128 query rows (16 per wave).  K tile: async->LDS double buffered.
// V tile: register-pipelined, stored transposed (Vt[dim][key]) for b128 frags.
__global__ __launch_bounds__(256) void attn_k(const bf16_t* __restrict__ qkv,
                                              bf16_t* __restrict__ attn,
                                              int B, int H, int T, int M) {
    const int Dh = 64;
    __shared__ __align__(16) bf16_t Ks[2][64 * 64];   // [key][dim]
    __shared__ __align__(16) bf16_t Vt[2][64 * 64];   // [dim][key]
    __shared__ __align__(16) bf16_t Ps[8 * 16 * 64];

    const int tid = threadIdx.x, wave = tid >> 5, lane = tid & 31;
    const int l16 = lane & 15, lhalf = lane >> 4;
    const int bh = blockIdx.y;
    const int b = bh / H, h = bh % H;
    const int qrow = blockIdx.x * 128 + wave * 16;    // wave's first query (t index)
    const int sr = tid >> 3, sc = tid & 7;            // stage coords (8 uint4/row)

    const bf16_t* qBase = qkv + (long)h * M * Dh;
    const bf16_t* kBase = qkv + (long)(H + h) * M * Dh;
    const bf16_t* vBase = qkv + (long)(2 * H + h) * M * Dh;

    // Q fragments: 16 rows x 64 dims = two 16x32 A-frags (held in VGPRs)
    v16bf qa[2];
    {
        long qoff = ((long)(b * T + qrow + l16)) * Dh;
#pragma unroll
        for (int i = 0; i < 2; i++) {
            const bf16_t* qp = qBase + qoff + i * 32 + lhalf * 8;
            ((uint4*)&qa[i])[0] = *(const uint4*)qp;
            ((uint4*)&qa[i])[1] = *(const uint4*)(qp + 16);
        }
    }

    v8f o[4];
    float mrow[8], lrow[8];
#pragma unroll
    for (int j = 0; j < 4; j++)
#pragma unroll
        for (int e = 0; e < 8; e++) o[j][e] = 0.f;
#pragma unroll
    for (int v = 0; v < 8; v++) { mrow[v] = NEGBIG; lrow[v] = 0.f; }

    // ---- prologue: key-block 0 in flight
    uint4 gv[2];
#pragma unroll
    for (int i = 0; i < 2; i++) {
        int idx = tid + i * 256;
        long grow = ((long)(b * T + sr + i * 32)) * Dh;
        async_ld_b128(kBase + grow + sc * 8, &Ks[0][idx * 8]);
        gv[i] = ((const uint4*)(vBase + grow))[sc];
    }

    const int nkb = 2 * blockIdx.x + 2;               // causal: keys <= block end
    for (int kb = 0; kb < nkb; kb++) {
        const int cur = kb & 1;
        const int kbase = kb * 64;
        {   // transposed V store: V[r][8c+e] -> Vt[8c+e][r]
#pragma unroll
            for (int i = 0; i < 2; i++) {
                const bf16_t* hb = (const bf16_t*)&gv[i];
#pragma unroll
                for (int e = 0; e < 8; e++)
                    Vt[cur][(sc * 8 + e) * 64 + sr + i * 32] = hb[e];
            }
        }
        wait_async0();
        __syncthreads();

        if (kb + 1 < nkb) {                           // launch key-block kb+1
            const int nxt = (kb + 1) & 1;
#pragma unroll
            for (int i = 0; i < 2; i++) {
                int idx = tid + i * 256;
                long grow = ((long)(b * T + kbase + 64 + sr + i * 32)) * Dh;
                async_ld_b128(kBase + grow + sc * 8, &Ks[nxt][idx * 8]);
                gv[i] = ((const uint4*)(vBase + grow))[sc];
            }
        }

        // S = Q @ K^T: per K-half, load 4 B-frags into distinct regs, then 4 WMMA
        v8f s[4];
#pragma unroll
        for (int j = 0; j < 4; j++)
#pragma unroll
            for (int e = 0; e < 8; e++) s[j][e] = 0.f;
#pragma unroll
        for (int i = 0; i < 2; i++) {
            v16bf bk4[4];
#pragma unroll
            for (int j = 0; j < 4; j++) {
                // B[k=dim][n=key] = Ks[key][dim] -> contiguous in dim
                const bf16_t* kp = &Ks[cur][(j * 16 + l16) * 64 + i * 32 + lhalf * 8];
                ((uint4*)&bk4[j])[0] = *(const uint4*)kp;
                ((uint4*)&bk4[j])[1] = *(const uint4*)(kp + 16);
            }
#pragma unroll
            for (int j = 0; j < 4; j++)
                s[j] = __builtin_amdgcn_wmma_f32_16x16x32_bf16(
                    false, qa[i], false, bk4[j], (short)0, s[j], false, false);
        }

        // scale; causal mask only on diagonal blocks (wave-uniform branch)
        if (kbase + 63 > qrow) {
#pragma unroll
            for (int j = 0; j < 4; j++)
#pragma unroll
                for (int v = 0; v < 8; v++) {
                    float sv = s[j][v] * 0.125f;      // 1/sqrt(64)
                    int key = kbase + j * 16 + l16;
                    int qg = qrow + v + lhalf * 8;
                    s[j][v] = (key > qg) ? NEGBIG : sv;
                }
        } else {
#pragma unroll
            for (int j = 0; j < 4; j++)
#pragma unroll
                for (int v = 0; v < 8; v++) s[j][v] *= 0.125f;
        }

        // online softmax: row stats per vgpr-element, reduce across 16-lane half
        float tmax[8];
#pragma unroll
        for (int v = 0; v < 8; v++)
            tmax[v] = fmaxf(fmaxf(s[0][v], s[1][v]), fmaxf(s[2][v], s[3][v]));
#pragma unroll
        for (int mk = 1; mk < 16; mk <<= 1)
#pragma unroll
            for (int v = 0; v < 8; v++) tmax[v] = fmaxf(tmax[v], __shfl_xor(tmax[v], mk, 32));

        float alpha[8], rsum[8];
#pragma unroll
        for (int v = 0; v < 8; v++) {
            float mn = fmaxf(mrow[v], tmax[v]);
            alpha[v] = __expf(mrow[v] - mn);
            mrow[v] = mn;
            rsum[v] = 0.f;
        }
#pragma unroll
        for (int j = 0; j < 4; j++)
#pragma unroll
            for (int v = 0; v < 8; v++) {
                float p = __expf(s[j][v] - mrow[v]);
                s[j][v] = p;
                rsum[v] += p;
            }
#pragma unroll
        for (int mk = 1; mk < 16; mk <<= 1)
#pragma unroll
            for (int v = 0; v < 8; v++) rsum[v] += __shfl_xor(rsum[v], mk, 32);
#pragma unroll
        for (int v = 0; v < 8; v++) lrow[v] = lrow[v] * alpha[v] + rsum[v];
#pragma unroll
        for (int j = 0; j < 4; j++)
#pragma unroll
            for (int v = 0; v < 8; v++) o[j][v] *= alpha[v];

        // P: C-layout -> A-layout via per-wave LDS round trip
        bf16_t* pw = &Ps[wave * 16 * 64];
#pragma unroll
        for (int j = 0; j < 4; j++)
#pragma unroll
            for (int v = 0; v < 8; v++)
                pw[(v + lhalf * 8) * 64 + j * 16 + l16] = (bf16_t)s[j][v];

        v16bf pa[2];
#pragma unroll
        for (int i = 0; i < 2; i++) {
            const bf16_t* pp = &pw[l16 * 64 + i * 32 + lhalf * 8];
            ((uint4*)&pa[i])[0] = *(const uint4*)pp;
            ((uint4*)&pa[i])[1] = *(const uint4*)(pp + 16);
        }

        // O += P @ V: per K-half, 4 V-frags (Vt contiguous in key) then 4 WMMA
#pragma unroll
        for (int i = 0; i < 2; i++) {
            v16bf bv4[4];
#pragma unroll
            for (int j = 0; j < 4; j++) {
                const bf16_t* vp = &Vt[cur][(j * 16 + l16) * 64 + i * 32 + lhalf * 8];
                ((uint4*)&bv4[j])[0] = *(const uint4*)vp;
                ((uint4*)&bv4[j])[1] = *(const uint4*)(vp + 16);
            }
#pragma unroll
            for (int j = 0; j < 4; j++)
                o[j] = __builtin_amdgcn_wmma_f32_16x16x32_bf16(
                    false, pa[i], false, bv4[j], (short)0, o[j], false, false);
        }
    }

    // normalize + store into [b*T+t][h*64+d] bf16 (input to Wo GEMM)
#pragma unroll
    for (int j = 0; j < 4; j++)
#pragma unroll
        for (int v = 0; v < 8; v++) {
            int qg = qrow + v + lhalf * 8;
            float val = o[j][v] / lrow[v];
            attn[((long)(b * T + qg)) * 1024 + h * 64 + j * 16 + l16] = (bf16_t)val;
        }
}

// ---------------------------------------------------------------- launch
extern "C" void kernel_launch(void* const* d_in, const int* in_sizes, int n_in,
                              void* d_out, int out_size, void* d_ws, size_t ws_size,
                              hipStream_t stream) {
    (void)in_sizes; (void)n_in; (void)out_size; (void)ws_size;
    const float* x   = (const float*)d_in[0];
    const float* Wq  = (const float*)d_in[1];
    const float* Wk  = (const float*)d_in[2];
    const float* Wv  = (const float*)d_in[3];
    const float* Wo  = (const float*)d_in[4];
    const float* bo  = (const float*)d_in[5];
    const float* W1  = (const float*)d_in[6];
    const float* b1  = (const float*)d_in[7];
    const float* W2  = (const float*)d_in[8];
    const float* b2  = (const float*)d_in[9];
    const float* g1  = (const float*)d_in[10];
    const float* be1 = (const float*)d_in[11];
    const float* g2  = (const float*)d_in[12];
    const float* be2 = (const float*)d_in[13];

    const int  Bb = 4, T = 2048, C = 1024, Hh = 16, Dh = 64, FF = 4096;
    const long M = (long)Bb * T;      // 8192 rows

    char* ws = (char*)d_ws;
    bf16_t* h1   = (bf16_t*)ws;  ws += M * C * 2;
    bf16_t* wqkv = (bf16_t*)ws;  ws += 3L * Hh * C * Dh * 2;
    bf16_t* woB  = (bf16_t*)ws;  ws += (long)C * C * 2;
    bf16_t* w1B  = (bf16_t*)ws;  ws += (long)C * FF * 2;
    bf16_t* w2B  = (bf16_t*)ws;  ws += (long)FF * C * 2;
    bf16_t* qkv  = (bf16_t*)ws;  ws += 3L * Hh * M * Dh * 2;
    bf16_t* attn = (bf16_t*)ws;  ws += M * C * 2;
    float*  x2   = (float*)ws;   ws += M * C * 4;
    bf16_t* h2   = (bf16_t*)ws;  ws += M * C * 2;
    bf16_t* ffa  = (bf16_t*)ws;  // M*FF*2

    // weights -> bf16 ([q heads][k heads][v heads] contiguous for z-indexed GEMM)
    cast_k<<<2048, 256, 0, stream>>>(Wq, wqkv,                     (long)Hh * C * Dh);
    cast_k<<<2048, 256, 0, stream>>>(Wk, wqkv + 1L * Hh * C * Dh,  (long)Hh * C * Dh);
    cast_k<<<2048, 256, 0, stream>>>(Wv, wqkv + 2L * Hh * C * Dh,  (long)Hh * C * Dh);
    cast_k<<<2048, 256, 0, stream>>>(Wo, woB, (long)C * C);
    cast_k<<<4096, 256, 0, stream>>>(W1, w1B, (long)C * FF);
    cast_k<<<4096, 256, 0, stream>>>(W2, w2B, (long)FF * C);

    // LN1
    ln_k<<<dim3((unsigned)M), 256, 0, stream>>>(x, g1, be1, h1, C);

    // QKV: 48 z-slices (3 matrices x 16 heads), each h1[8192x1024] @ W[1024x64]
    gemm_bf16_k<0, 64><<<dim3(1, (unsigned)(M / 128), 3 * Hh), 256, 0, stream>>>(
        h1, wqkv, (int)M, Dh, C, nullptr, nullptr, nullptr, qkv,
        (long)C * Dh, (long)M * Dh);

    // causal flash attention
    attn_k<<<dim3(T / 128, Bb * Hh), 256, 0, stream>>>(qkv, attn, Bb, Hh, T, (int)M);

    // x2 = x + attn @ Wo + bo
    gemm_bf16_k<2, 128><<<dim3(C / 128, (unsigned)(M / 128)), 256, 0, stream>>>(
        attn, woB, (int)M, C, C, bo, x, x2, nullptr, 0, 0);

    // LN2
    ln_k<<<dim3((unsigned)M), 256, 0, stream>>>(x2, g2, be2, h2, C);

    // FF1: relu(h2 @ W1 + b1) -> bf16
    gemm_bf16_k<1, 128><<<dim3(FF / 128, (unsigned)(M / 128)), 256, 0, stream>>>(
        h2, w1B, (int)M, FF, C, b1, nullptr, nullptr, ffa, 0, 0);

    // out = x2 + ffa @ W2 + b2
    gemm_bf16_k<2, 128><<<dim3(C / 128, (unsigned)(M / 128)), 256, 0, stream>>>(
        ffa, w2B, (int)M, C, FF, b2, x2, (float*)d_out, nullptr, 0, 0);
}
